// CompressMultiheadAttention_19559281066305
// MI455X (gfx1250) — compile-verified, matrix operand order checked by reading, not run on hardware
//
#include <hip/hip_runtime.h>
#include <hip/hip_bf16.h>

typedef __attribute__((ext_vector_type(16))) __bf16 v16bf;
typedef __attribute__((ext_vector_type(8)))  float  v8f;

// ---------- helpers ----------
__device__ __forceinline__ unsigned short f2bf(float f) {
    unsigned int u = __builtin_bit_cast(unsigned int, f);
    unsigned int r = (u + 0x7fffu + ((u >> 16) & 1u)) >> 16;
    return (unsigned short)r;
}
__device__ __forceinline__ __bf16 bfbits(unsigned short u) {
    return __builtin_bit_cast(__bf16, u);
}

// 16x32 bf16 WMMA fragment load (A layout; B uses same math on transposed src).
// Lane l holds row l%16; VGPR vg holds k-pair (vg/4)*16 + (l/16)*8 + (vg%4)*2.
// Two contiguous 16B groups per lane -> compiler emits ds_load_b128 pairs.
__device__ __forceinline__ v16bf load_frag(const unsigned short* __restrict__ p,
                                           int ld, int lane) {
    const int r = lane & 15;
    const int half = (lane >> 4) & 1;
    const unsigned short* rp = p + r * ld + half * 8;
    v16bf f;
#pragma unroll
    for (int vg = 0; vg < 8; ++vg) {
        int k = ((vg >> 2) << 4) + ((vg & 3) << 1);
        f[2 * vg]     = bfbits(rp[k]);
        f[2 * vg + 1] = bfbits(rp[k + 1]);
    }
    return f;
}

__device__ __forceinline__ v8f wmma_bf16(v16bf a, v16bf b, v8f c) {
    return __builtin_amdgcn_wmma_f32_16x16x32_bf16(false, a, false, b,
                                                   (short)0, c, false, false);
}

// Async copy 16B global -> LDS (ASYNCcnt-tracked). lds_off = LDS byte address
// (= low 32 bits of the generic pointer to a __shared__ object).
__device__ __forceinline__ void async_b128(unsigned int lds_off, const void* g) {
    asm volatile("global_load_async_to_lds_b128 %0, %1, off"
                 :: "v"(lds_off), "v"(g) : "memory");
}
__device__ __forceinline__ unsigned int lds_off(const void* p) {
    return (unsigned int)(unsigned long long)p;
}

// ---------- kernel 0a: f32 -> bf16 convert (optionally scaled) ----------
__global__ void cvt_bf16(const float* __restrict__ src,
                         unsigned short* __restrict__ dst, int n, float scale) {
    int i = blockIdx.x * blockDim.x + threadIdx.x;
    int stride = gridDim.x * blockDim.x;
    for (; i < n; i += stride) dst[i] = f2bf(src[i] * scale);
}

// ---------- kernel 0b: W (f32, k x n) -> W^T (bf16, n x k), 1024x1024 ----------
__global__ __launch_bounds__(256) void tr_cvt_bf16(const float* __restrict__ src,
                                                   unsigned short* __restrict__ dst) {
    __shared__ unsigned short t[32][33];
    const int bx = blockIdx.x * 32, by = blockIdx.y * 32;
    const int x = threadIdx.x & 31, y0 = threadIdx.x >> 5;
#pragma unroll
    for (int yy = y0; yy < 32; yy += 8)
        t[yy][x] = f2bf(src[(size_t)(by + yy) * 1024 + bx + x]);
    __syncthreads();
#pragma unroll
    for (int yy = y0; yy < 32; yy += 8)
        dst[(size_t)(bx + yy) * 1024 + by + x] = t[x][yy];
}

// ---------- GEMM: A (bf16, (B*S) x 1024) @ BT^T + bias ----------
// BT is the 1024x1024 weight stored TRANSPOSED (n,k). Block tile 128x128,
// 8 waves x (2x4) 16x16 subtiles, K-step 32, double-buffered async staging.
template <bool HEADSPLIT>
__global__ __launch_bounds__(256) void gemm_bf16(
    const unsigned short* __restrict__ Ag, const unsigned short* __restrict__ BTg,
    const float* __restrict__ bias, unsigned short* __restrict__ dsth,
    float* __restrict__ dstf) {
    constexpr int LDA = 40;  // 80B pitch (16B multiple)
    __shared__ alignas(16) unsigned short As[2][128 * LDA];
    __shared__ alignas(16) unsigned short Bs[2][128 * LDA];
    const int tid = threadIdx.x, lane = tid & 31, wave = tid >> 5;
    const int r0 = blockIdx.x * 128, c0 = blockIdx.y * 128;
    const int rw = (wave & 3) * 2;
    const int cw = (wave >> 2) * 4;

    v8f acc[2][4];
#pragma unroll
    for (int t = 0; t < 2; ++t)
#pragma unroll
        for (int cc = 0; cc < 4; ++cc) acc[t][cc] = v8f{};

    auto stage = [&](int buf, int k0) {  // 4 async b128 per thread
#pragma unroll
        for (int j = 0; j < 2; ++j) {
            int e = tid + j * 256;
            int i = e >> 2, seg = (e & 3) * 8;
            async_b128(lds_off(&As[buf][i * LDA + seg]),
                       Ag + (size_t)(r0 + i) * 1024 + k0 + seg);
        }
#pragma unroll
        for (int j = 0; j < 2; ++j) {
            int e = tid + j * 256;
            int nn = e >> 2, seg = (e & 3) * 8;
            async_b128(lds_off(&Bs[buf][nn * LDA + seg]),
                       BTg + (size_t)(c0 + nn) * 1024 + k0 + seg);
        }
    };

    stage(0, 0);
    int buf = 0;
    for (int k0 = 0; k0 < 1024; k0 += 32) {
        if (k0 + 32 < 1024) {
            stage(buf ^ 1, k0 + 32);
            asm volatile("s_wait_asynccnt 0x4" ::: "memory");
        } else {
            asm volatile("s_wait_asynccnt 0x0" ::: "memory");
        }
        __syncthreads();
        v16bf a0 = load_frag(&As[buf][(rw + 0) * 16 * LDA], LDA, lane);
        v16bf a1 = load_frag(&As[buf][(rw + 1) * 16 * LDA], LDA, lane);
#pragma unroll
        for (int cc = 0; cc < 4; ++cc) {
            v16bf bb = load_frag(&Bs[buf][(cw + cc) * 16 * LDA], LDA, lane);
            acc[0][cc] = wmma_bf16(a0, bb, acc[0][cc]);
            acc[1][cc] = wmma_bf16(a1, bb, acc[1][cc]);
        }
        __syncthreads();
        buf ^= 1;
    }
    const int n = lane & 15, half = lane >> 4;
#pragma unroll
    for (int t = 0; t < 2; ++t)
#pragma unroll
        for (int cc = 0; cc < 4; ++cc)
#pragma unroll
            for (int j = 0; j < 8; ++j) {
                int r = r0 + (rw + t) * 16 + j + 8 * half;
                int c = c0 + (cw + cc) * 16 + n;
                float v = acc[t][cc][j] + bias[c];
                if (HEADSPLIT) {
                    int b = r >> 12, s = r & 4095;
                    int h = c >> 6, d = c & 63;
                    dsth[(size_t)(((b * 16 + h) * 4096) + s) * 64 + d] = f2bf(v);
                } else {
                    dstf[(size_t)r * 1024 + c] = v;
                }
            }
}

// ---------- attention pass 1 (split-S partials) ----------
// Partial of step1 = softmax((c/sqrt(DH)) @ K^T) @ V over one S-split of 1024.
// grid (2 m-halves, 4 s-splits, 64 bh) = 512 blocks, 8 waves, wave = 16 m-rows.
// Emits unnormalized O + per-row (max, sum) stats for the combine kernel.
__global__ __launch_bounds__(256) void attn1_part(
    const unsigned short* __restrict__ Kg, const unsigned short* __restrict__ Vg,
    const unsigned short* __restrict__ Cbf, const unsigned char* __restrict__ msk,
    float* __restrict__ Op, float* __restrict__ St) {
    constexpr int KLD = 72, PLD = 72;  // 144B pitch
    __shared__ alignas(16) unsigned short Ks[64 * KLD];
    __shared__ alignas(16) unsigned short Vt[64 * KLD];
    __shared__ alignas(16) unsigned short Ps[8][16 * PLD];
    const int tid = threadIdx.x, lane = tid & 31, wave = tid >> 5;
    const int split = blockIdx.y;
    const int bh = blockIdx.z, b = bh >> 4, h = bh & 15;
    const int m0 = blockIdx.x * 128 + wave * 16;
    const unsigned short* Kbh = Kg + (size_t)bh * 4096 * 64;
    const unsigned short* Vbh = Vg + (size_t)bh * 4096 * 64;
    const unsigned short* Ch  = Cbf + (size_t)h * 256 * 64;
    const int n = lane & 15, half = lane >> 4;

    v16bf ca0 = load_frag(Ch + m0 * 64 + 0, 64, lane);
    v16bf ca1 = load_frag(Ch + m0 * 64 + 32, 64, lane);
    v8f O[4] = {v8f{}, v8f{}, v8f{}, v8f{}};
    float rmax[8], rsum[8];
#pragma unroll
    for (int j = 0; j < 8; ++j) { rmax[j] = -1e30f; rsum[j] = 0.f; }

    const int sbeg = split * 1024;
    for (int sb = sbeg; sb < sbeg + 1024; sb += 64) {
#pragma unroll
        for (int j = 0; j < 2; ++j) {  // K chunk: async, row-major
            int e = tid + j * 256;
            int sl = e >> 2, seg = (e & 3) * 8;
            async_b128(lds_off(&Ks[sl * KLD + seg]),
                       Kbh + (size_t)(sb + sl) * 64 + seg);
        }
#pragma unroll
        for (int j = 0; j < 16; ++j) {  // V chunk: manual transpose
            int e = tid + j * 256;
            int sl = e >> 6, d = e & 63;
            Vt[d * KLD + sl] = Vbh[(size_t)(sb + sl) * 64 + d];
        }
        asm volatile("s_wait_asynccnt 0x0" ::: "memory");
        __syncthreads();
        v8f sc[4] = {v8f{}, v8f{}, v8f{}, v8f{}};
#pragma unroll
        for (int st = 0; st < 4; ++st) {
            v16bf bk0 = load_frag(Ks + (st * 16) * KLD + 0, KLD, lane);
            v16bf bk1 = load_frag(Ks + (st * 16) * KLD + 32, KLD, lane);
            sc[st] = wmma_bf16(ca0, bk0, sc[st]);
            sc[st] = wmma_bf16(ca1, bk1, sc[st]);
        }
#pragma unroll
        for (int st = 0; st < 4; ++st) {  // key mask (column == lane's n)
            int scol = sb + st * 16 + n;
            if (!msk[(size_t)b * 4096 + scol]) {
#pragma unroll
                for (int j = 0; j < 8; ++j) sc[st][j] = -1e30f;
            }
        }
#pragma unroll
        for (int j = 0; j < 8; ++j) {  // online softmax, row j+8*half
            float mx = fmaxf(fmaxf(sc[0][j], sc[1][j]), fmaxf(sc[2][j], sc[3][j]));
#pragma unroll
            for (int xm = 1; xm <= 8; xm <<= 1) mx = fmaxf(mx, __shfl_xor(mx, xm, 32));
            float nm = fmaxf(rmax[j], mx);
            float corr = __expf(rmax[j] - nm);
            rmax[j] = nm;
            rsum[j] *= corr;
#pragma unroll
            for (int dt = 0; dt < 4; ++dt) O[dt][j] *= corr;
            float ps = 0.f;
#pragma unroll
            for (int st = 0; st < 4; ++st) {
                float p = __expf(sc[st][j] - nm);
                sc[st][j] = p;
                ps += p;
            }
#pragma unroll
            for (int xm = 1; xm <= 8; xm <<= 1) ps += __shfl_xor(ps, xm, 32);
            rsum[j] += ps;
        }
#pragma unroll
        for (int st = 0; st < 4; ++st)  // stage P (bf16) per-wave in LDS
#pragma unroll
            for (int j = 0; j < 8; ++j)
                Ps[wave][(j + 8 * half) * PLD + st * 16 + n] = f2bf(sc[st][j]);
        asm volatile("s_wait_dscnt 0" ::: "memory");
        v16bf pa0 = load_frag(&Ps[wave][0], PLD, lane);
        v16bf pa1 = load_frag(&Ps[wave][32], PLD, lane);
#pragma unroll
        for (int dt = 0; dt < 4; ++dt) {
            v16bf bv0 = load_frag(Vt + (dt * 16) * KLD + 0, KLD, lane);
            v16bf bv1 = load_frag(Vt + (dt * 16) * KLD + 32, KLD, lane);
            O[dt] = wmma_bf16(pa0, bv0, O[dt]);
            O[dt] = wmma_bf16(pa1, bv1, O[dt]);
        }
        __syncthreads();
    }
    const size_t base = (size_t)(bh * 4 + split) * 256;
#pragma unroll
    for (int dt = 0; dt < 4; ++dt)
#pragma unroll
        for (int j = 0; j < 8; ++j) {
            int d = dt * 16 + n;
            int m = m0 + j + 8 * half;
            Op[(base + m) * 64 + d] = O[dt][j];
        }
    if (n == 0) {
#pragma unroll
        for (int j = 0; j < 8; ++j) {
            int m = m0 + j + 8 * half;
            St[(base + m) * 2 + 0] = rmax[j];
            St[(base + m) * 2 + 1] = rsum[j];
        }
    }
}

// ---------- attention pass 1 combine ----------
// Merge 4 split partials with log-sum-exp weights; store step1^T bf16.
// grid (64 bh, 64 m-quads), block 256 = 4 rows x 64 d.
__global__ __launch_bounds__(256) void attn1_combine(
    const float* __restrict__ Op, const float* __restrict__ St,
    unsigned short* __restrict__ S1T) {
    const int bh = blockIdx.x;
    const int m = blockIdx.y * 4 + (threadIdx.x >> 6);
    const int d = threadIdx.x & 63;
    float mi[4], si[4], M = -1e30f;
#pragma unroll
    for (int i = 0; i < 4; ++i) {
        size_t row = (size_t)(bh * 4 + i) * 256 + m;
        mi[i] = St[row * 2 + 0];
        si[i] = St[row * 2 + 1];
        M = fmaxf(M, mi[i]);
    }
    float denom = 0.f, acc = 0.f;
#pragma unroll
    for (int i = 0; i < 4; ++i) {
        size_t row = (size_t)(bh * 4 + i) * 256 + m;
        float w = __expf(mi[i] - M);
        denom += w * si[i];
        acc += w * Op[row * 64 + d];
    }
    S1T[(size_t)bh * 64 * 256 + (size_t)d * 256 + m] = f2bf(acc / denom);
}

// ---------- attention pass 2 ----------
// heads = softmax(Q @ (c/sqrt(DH))^T) @ step1, M=256 in registers.
// grid (32 s-blocks of 128, 64 bh), 8 waves; wave owns 16 s-rows.
__global__ __launch_bounds__(256) void attn2(
    const unsigned short* __restrict__ Qg, const unsigned short* __restrict__ Cbf,
    const unsigned short* __restrict__ S1T, unsigned short* __restrict__ Hd) {
    constexpr int QLD = 72, PLD = 40;
    __shared__ alignas(16) unsigned short Qs[128 * QLD];
    __shared__ alignas(16) unsigned short Ps[8][16 * PLD];
    const int tid = threadIdx.x, lane = tid & 31, wave = tid >> 5;
    const int bh = blockIdx.y, b = bh >> 4, h = bh & 15;
    const int s0 = blockIdx.x * 128;
    const unsigned short* Qbh = Qg + (size_t)bh * 4096 * 64;
    const unsigned short* Ch  = Cbf + (size_t)h * 256 * 64;
    const unsigned short* S1  = S1T + (size_t)bh * 64 * 256;
    const int n = lane & 15, half = lane >> 4;

#pragma unroll
    for (int j = 0; j < 4; ++j) {  // stage Q 128x64 via async b128
        int e = tid + j * 256;
        int sl = e >> 3, seg = (e & 7) * 8;
        async_b128(lds_off(&Qs[sl * QLD + seg]),
                   Qbh + (size_t)(s0 + sl) * 64 + seg);
    }
    asm volatile("s_wait_asynccnt 0x0" ::: "memory");
    __syncthreads();
    v16bf qa0 = load_frag(Qs + (wave * 16) * QLD + 0, QLD, lane);
    v16bf qa1 = load_frag(Qs + (wave * 16) * QLD + 32, QLD, lane);
    v8f sc[16];
#pragma unroll
    for (int mt = 0; mt < 16; ++mt) {
        v16bf b0 = load_frag(Ch + (mt * 16) * 64 + 0, 64, lane);
        v16bf b1 = load_frag(Ch + (mt * 16) * 64 + 32, 64, lane);
        v8f a = {};
        a = wmma_bf16(qa0, b0, a);
        a = wmma_bf16(qa1, b1, a);
        sc[mt] = a;
    }
    float rsum[8];
#pragma unroll
    for (int j = 0; j < 8; ++j) {
        float mx = -1e30f;
#pragma unroll
        for (int mt = 0; mt < 16; ++mt) mx = fmaxf(mx, sc[mt][j]);
#pragma unroll
        for (int xm = 1; xm <= 8; xm <<= 1) mx = fmaxf(mx, __shfl_xor(mx, xm, 32));
        float s = 0.f;
#pragma unroll
        for (int mt = 0; mt < 16; ++mt) {
            float p = __expf(sc[mt][j] - mx);
            sc[mt][j] = p;
            s += p;
        }
#pragma unroll
        for (int xm = 1; xm <= 8; xm <<= 1) s += __shfl_xor(s, xm, 32);
        rsum[j] = s;
    }
    v8f O[4] = {v8f{}, v8f{}, v8f{}, v8f{}};
    for (int mc = 0; mc < 8; ++mc) {  // P @ step1, 32-wide m chunks
#pragma unroll
        for (int t = 0; t < 2; ++t) {
            int mt = mc * 2 + t;
#pragma unroll
            for (int j = 0; j < 8; ++j)
                Ps[wave][(j + 8 * half) * PLD + t * 16 + n] = f2bf(sc[mt][j]);
        }
        asm volatile("s_wait_dscnt 0" ::: "memory");
        v16bf pa = load_frag(&Ps[wave][0], PLD, lane);
#pragma unroll
        for (int dt = 0; dt < 4; ++dt) {
            v16bf bb = load_frag(S1 + (dt * 16) * 256 + mc * 32, 256, lane);
            O[dt] = wmma_bf16(pa, bb, O[dt]);
        }
    }
#pragma unroll
    for (int dt = 0; dt < 4; ++dt)
#pragma unroll
        for (int j = 0; j < 8; ++j) {
            float v = O[dt][j] / rsum[j];
            int s = s0 + wave * 16 + j + 8 * half;
            int col = h * 64 + dt * 16 + n;
            Hd[(size_t)(b * 4096 + s) * 1024 + col] = f2bf(v);
        }
}

// ---------- host ----------
extern "C" void kernel_launch(void* const* d_in, const int* in_sizes, int n_in,
                              void* d_out, int out_size, void* d_ws, size_t ws_size,
                              hipStream_t stream) {
    (void)in_sizes; (void)n_in; (void)out_size; (void)ws_size;
    const float* queries = (const float*)d_in[0];
    const unsigned char* msk = (const unsigned char*)d_in[1];
    const float* Wq = (const float*)d_in[2];
    const float* bq = (const float*)d_in[3];
    const float* Wk = (const float*)d_in[4];
    const float* bk = (const float*)d_in[5];
    const float* Wv = (const float*)d_in[6];
    const float* bv = (const float*)d_in[7];
    const float* Wo = (const float*)d_in[8];
    const float* bo = (const float*)d_in[9];
    const float* c  = (const float*)d_in[10];
    float* out = (float*)d_out;

    char* ws = (char*)d_ws;
    size_t off = 0;
    auto alloc = [&](size_t bytes) {
        void* p = ws + off;
        off = (off + bytes + 255) & ~(size_t)255;
        return p;
    };
    const size_t DD = 1024 * 1024;
    const size_t BSD = (size_t)4 * 4096 * 1024;
    unsigned short* x_bf  = (unsigned short*)alloc(BSD * 2);
    unsigned short* wqT   = (unsigned short*)alloc(DD * 2);
    unsigned short* wkT   = (unsigned short*)alloc(DD * 2);
    unsigned short* wvT   = (unsigned short*)alloc(DD * 2);
    unsigned short* woT   = (unsigned short*)alloc(DD * 2);
    unsigned short* c_bf  = (unsigned short*)alloc((size_t)16 * 256 * 64 * 2);
    unsigned short* q_ws  = (unsigned short*)alloc((size_t)64 * 4096 * 64 * 2);
    unsigned short* k_ws  = (unsigned short*)alloc((size_t)64 * 4096 * 64 * 2);
    unsigned short* v_ws  = (unsigned short*)alloc((size_t)64 * 4096 * 64 * 2);
    unsigned short* s1t   = (unsigned short*)alloc((size_t)64 * 64 * 256 * 2);
    unsigned short* heads = (unsigned short*)alloc(BSD * 2);
    float* o_part = (float*)alloc((size_t)64 * 4 * 256 * 64 * 4);  // split partials
    float* stats  = (float*)alloc((size_t)64 * 4 * 256 * 2 * 4);   // (max,sum)

    cvt_bf16<<<4096, 256, 0, stream>>>(queries, x_bf, (int)BSD, 1.0f);
    cvt_bf16<<<1024, 256, 0, stream>>>(c, c_bf, 16 * 256 * 64, 0.125f); // 1/sqrt(64)
    dim3 tg(32, 32);
    tr_cvt_bf16<<<tg, 256, 0, stream>>>(Wq, wqT);
    tr_cvt_bf16<<<tg, 256, 0, stream>>>(Wk, wkT);
    tr_cvt_bf16<<<tg, 256, 0, stream>>>(Wv, wvT);
    tr_cvt_bf16<<<tg, 256, 0, stream>>>(Wo, woT);

    dim3 gg(128, 8);
    gemm_bf16<true><<<gg, 256, 0, stream>>>(x_bf, wqT, bq, q_ws, nullptr);
    gemm_bf16<true><<<gg, 256, 0, stream>>>(x_bf, wkT, bk, k_ws, nullptr);
    gemm_bf16<true><<<gg, 256, 0, stream>>>(x_bf, wvT, bv, v_ws, nullptr);

    attn1_part<<<dim3(2, 4, 64), 256, 0, stream>>>(k_ws, v_ws, c_bf, msk,
                                                   o_part, stats);
    attn1_combine<<<dim3(64, 64), 256, 0, stream>>>(o_part, stats, s1t);
    attn2<<<dim3(32, 64), 256, 0, stream>>>(q_ws, c_bf, s1t, heads);

    gemm_bf16<false><<<gg, 256, 0, stream>>>(heads, woT, bo, nullptr, out);
}
